// Decoder_31756988187337
// MI455X (gfx1250) — compile-verified, matrix operand order checked by reading, not run on hardware
//
#include <hip/hip_runtime.h>

typedef __attribute__((ext_vector_type(16))) _Float16 v16h;
typedef __attribute__((ext_vector_type(8)))  float    v8f;

#define EPS_F 1e-9f
#define ROWS 4096          // S*B
#define KH   64            // 60 padded to 64
#define NH   512
#define NB   160
#define NP   12288         // 3*64*64
#define GROUPS 192         // NP / 64  (= C*Hd)

// ---------------------------------------------------------------------------
// Fragment loaders (wave32, ISA 7.12.2 layouts)
// ---------------------------------------------------------------------------

// A fragment 16x32 f16 from a row-major [*, ld] matrix at (row0, k0).
// Lane l(0-15): M=l, K = k0 + {0..7, 16..23}; lane l+16: K = k0 + {8..15, 24..31}.
__device__ inline v16h load_a_rowmajor(const _Float16* A, int ld, int row0, int k0) {
    int lane = threadIdx.x & 31;
    int m  = lane & 15;
    int hi = lane >> 4;
    const _Float16* p = A + (size_t)(row0 + m) * ld + k0 + hi * 8;
    v16h a;
#pragma unroll
    for (int j = 0; j < 8; ++j) a[j]     = p[j];
#pragma unroll
    for (int j = 0; j < 8; ++j) a[8 + j] = p[16 + j];
    return a;
}

// B fragment 32x16 f16 from fragment-packed weights: 512 halfs per (kt,nt)
// fragment, 16 contiguous halfs per lane -> two global_load_b128 per lane.
__device__ inline v16h load_b_packed(const _Float16* Bp, int nTiles, int kt, int nt) {
    int lane = threadIdx.x & 31;
    const _Float16* p = Bp + (((size_t)kt * nTiles + nt) * 32 + lane) * 16;
    v16h b;
#pragma unroll
    for (int j = 0; j < 16; ++j) b[j] = p[j];
    return b;
}

__device__ inline v8f wmma_f16(v16h a, v16h b, v8f c) {
    return __builtin_amdgcn_wmma_f32_16x16x32_f16(false, a, false, b, (short)0, c,
                                                  false, false);
}

// ---------------------------------------------------------------------------
// Prep kernels: run once per launch, tiny cost (~4 MB of weight repack).
// ---------------------------------------------------------------------------

// Pack f32 weight [Kreal x N] into f16 B-fragment layout padded to Kpad.
__global__ void pack_b_f16(const float* __restrict__ W, _Float16* __restrict__ out,
                           int Kreal, int Kpad, int N) {
    int tid = blockIdx.x * blockDim.x + threadIdx.x;
    int total = Kpad * N;
    if (tid >= total) return;
    int j    = tid & 15;
    int lane = (tid >> 4) & 31;
    int frag = tid >> 9;              // 512 halfs per fragment
    int nTiles = N >> 4;
    int kt = frag / nTiles;
    int nt = frag - kt * nTiles;
    int hi = lane >> 4;
    int n  = (lane & 15) + nt * 16;
    int k  = kt * 32 + ((j < 8) ? (hi * 8 + j) : (16 + hi * 8 + (j - 8)));
    float v = (k < Kreal) ? W[(size_t)k * N + n] : 0.0f;
    out[tid] = (_Float16)v;
}

// Build padded input X f16 [ROWS x 64]: cols 0..9 digits, 10..59 styles, 60..63 zero.
__global__ void pack_x_f16(const float* __restrict__ digits,
                           const float* __restrict__ styles,
                           _Float16* __restrict__ X) {
    int tid = blockIdx.x * blockDim.x + threadIdx.x;
    if (tid >= ROWS * KH) return;
    int r = tid >> 6;
    int c = tid & 63;
    float v = 0.0f;
    if (c < 10)       v = digits[r * 10 + c];
    else if (c < 60)  v = styles[r * 50 + (c - 10)];
    X[tid] = (_Float16)v;
}

// ---------------------------------------------------------------------------
// GEMM kernels: one wave per 16x16 tile (wave32), WMMA f16 -> f32 acc.
// ---------------------------------------------------------------------------

__global__ __launch_bounds__(256)
void gemm_bias_relu_f16(const _Float16* __restrict__ A, const _Float16* __restrict__ Bp,
                        const float* __restrict__ bias, _Float16* __restrict__ Out,
                        int M, int K, int N) {
    int wid  = (blockIdx.x * blockDim.x + threadIdx.x) >> 5;
    int lane = threadIdx.x & 31;
    int nTiles = N >> 4;
    int mt = wid / nTiles;
    int nt = wid - mt * nTiles;
    if (mt * 16 >= M) return;

    v8f acc = {};
    int kTiles = K >> 5;
    for (int kt = 0; kt < kTiles; ++kt) {
        v16h a = load_a_rowmajor(A, K, mt * 16, kt * 32);
        v16h b = load_b_packed(Bp, nTiles, kt, nt);
        acc = wmma_f16(a, b, acc);
    }
    int n  = nt * 16 + (lane & 15);
    int hi = lane >> 4;
    float bv = bias[n];
#pragma unroll
    for (int r = 0; r < 8; ++r) {
        int m = mt * 16 + r + hi * 8;
        float v = acc[r] + bv;
        v = v > 0.0f ? v : 0.0f;
        Out[(size_t)m * N + n] = (_Float16)v;
    }
}

__global__ __launch_bounds__(256)
void gemm_bias_selu_f16(const _Float16* __restrict__ A, const _Float16* __restrict__ Bp,
                        const float* __restrict__ bias, _Float16* __restrict__ Out,
                        int M, int K, int N) {
    int wid  = (blockIdx.x * blockDim.x + threadIdx.x) >> 5;
    int lane = threadIdx.x & 31;
    int nTiles = N >> 4;
    int mt = wid / nTiles;
    int nt = wid - mt * nTiles;
    if (mt * 16 >= M) return;

    v8f acc = {};
    int kTiles = K >> 5;
    for (int kt = 0; kt < kTiles; ++kt) {
        v16h a = load_a_rowmajor(A, K, mt * 16, kt * 32);
        v16h b = load_b_packed(Bp, nTiles, kt, nt);
        acc = wmma_f16(a, b, acc);
    }
    const float kScale = 1.0507009873554805f;
    const float kAlpha = 1.6732632423543772f;
    int n  = nt * 16 + (lane & 15);
    int hi = lane >> 4;
    float bv = bias[n];
#pragma unroll
    for (int r = 0; r < 8; ++r) {
        int m = mt * 16 + r + hi * 8;
        float x = acc[r] + bv;
        float v = (x > 0.0f) ? (kScale * x) : (kScale * kAlpha * (__expf(x) - 1.0f));
        Out[(size_t)m * N + n] = (_Float16)v;
    }
}

// Fused: x_hat = sigmoid(t @ W2 + b2); loss = -(log(x+eps)*img +
// log(1-x+eps)*(1-img)).sum over Wd=64. One wave owns a 16x64 tile
// (4 N-tiles = one full Wd group) so the reduction is wave-local.
//
// Images are streamed exactly once (201 MB = the HBM roofline term), so:
//  - they are loaded with NT temporal hint to keep L2 free for W2/t reuse,
//  - the loads are issued BEFORE the WMMA K-loop so HBM latency overlaps
//    the 20-WMMA compute chain instead of serializing after it.
__global__ __launch_bounds__(256)
void gemm_sigmoid_bce(const _Float16* __restrict__ T, const _Float16* __restrict__ Bp,
                      const float* __restrict__ bias, const float* __restrict__ images,
                      float* __restrict__ out, int M, int K, int N) {
    int wid  = (blockIdx.x * blockDim.x + threadIdx.x) >> 5;
    int lane = threadIdx.x & 31;
    int groups = N >> 6;                 // 192
    int mt = wid / groups;
    int g  = wid - mt * groups;
    if (mt * 16 >= M) return;

    int hi = lane >> 4;
    int nl = lane & 15;

    // ---- issue streaming image loads first (overlap with WMMA below) ----
    float img[4][8];
#pragma unroll
    for (int t = 0; t < 4; ++t) {
        int n = g * 64 + t * 16 + nl;
        const float* ip = images + (size_t)(mt * 16 + hi * 8) * N + n;
#pragma unroll
        for (int r = 0; r < 8; ++r)
            img[t][r] = __builtin_nontemporal_load(ip + (size_t)r * N);
    }
    float bv[4];
#pragma unroll
    for (int t = 0; t < 4; ++t) bv[t] = bias[g * 64 + t * 16 + nl];

    // ---- WMMA K-loop: A fragment reused across 4 N-tiles ----
    v8f acc[4] = {v8f{}, v8f{}, v8f{}, v8f{}};
    int kTiles = K >> 5;                 // 5
    int nTiles = N >> 4;                 // 768
    for (int kt = 0; kt < kTiles; ++kt) {
        v16h a = load_a_rowmajor(T, K, mt * 16, kt * 32);
#pragma unroll
        for (int t = 0; t < 4; ++t) {
            v16h b = load_b_packed(Bp, nTiles, kt, g * 4 + t);
            acc[t] = wmma_f16(a, b, acc[t]);
        }
    }

    // ---- sigmoid + eps-guarded BCE, accumulate per row ----
    float rowsum[8];
#pragma unroll
    for (int r = 0; r < 8; ++r) rowsum[r] = 0.0f;

#pragma unroll
    for (int t = 0; t < 4; ++t) {
#pragma unroll
        for (int r = 0; r < 8; ++r) {
            float z = acc[t][r] + bv[t];
            float p = 1.0f / (1.0f + __expf(-z));          // sigmoid
            float im = img[t][r];
            rowsum[r] += __logf(p + EPS_F) * im
                       + __logf(1.0f - p + EPS_F) * (1.0f - im);
        }
    }

    // Reduce across the 16 lanes of each half-wave (cols within the group).
#pragma unroll
    for (int r = 0; r < 8; ++r) {
        float s = rowsum[r];
#pragma unroll
        for (int off = 1; off < 16; off <<= 1) s += __shfl_xor(s, off, 32);
        rowsum[r] = s;
    }
    if (nl == 0) {
#pragma unroll
        for (int r = 0; r < 8; ++r) {
            int m = mt * 16 + r + hi * 8;
            out[(size_t)m * groups + g] = -rowsum[r];
        }
    }
}

// ---------------------------------------------------------------------------
// Launch
// ---------------------------------------------------------------------------
extern "C" void kernel_launch(void* const* d_in, const int* in_sizes, int n_in,
                              void* d_out, int out_size, void* d_ws, size_t ws_size,
                              hipStream_t stream) {
    const float* digits = (const float*)d_in[0];   // [8,512,10]
    const float* styles = (const float*)d_in[1];   // [8,512,50]
    const float* images = (const float*)d_in[2];   // [8,512,3,64,64]
    const float* W_h    = (const float*)d_in[3];   // [60,512]
    const float* b_h    = (const float*)d_in[4];   // [512]
    const float* W1     = (const float*)d_in[5];   // [512,160]
    const float* b1     = (const float*)d_in[6];   // [160]
    const float* W2     = (const float*)d_in[7];   // [160,12288]
    const float* b2     = (const float*)d_in[8];   // [12288]
    float* out = (float*)d_out;                    // [8,512,3,64] = 4096*192

    // Workspace carve-up (f16), ~9.7 MB total.
    char* w = (char*)d_ws;
    _Float16* Xp  = (_Float16*)w;  w += (size_t)ROWS * KH * 2;   // padded input
    _Float16* WhP = (_Float16*)w;  w += (size_t)KH * NH * 2;     // packed W_h
    _Float16* W1P = (_Float16*)w;  w += (size_t)NH * NB * 2;     // packed W1
    _Float16* W2P = (_Float16*)w;  w += (size_t)NB * NP * 2;     // packed W2
    _Float16* Hb  = (_Float16*)w;  w += (size_t)ROWS * NH * 2;   // h (row-major)
    _Float16* Tb  = (_Float16*)w;  w += (size_t)ROWS * NB * 2;   // t (row-major)

    const int TPB = 256;
    // Prep: pack input + weights to f16 fragment layouts.
    pack_x_f16<<<(ROWS * KH + TPB - 1) / TPB, TPB, 0, stream>>>(digits, styles, Xp);
    pack_b_f16<<<(KH * NH + TPB - 1) / TPB, TPB, 0, stream>>>(W_h, WhP, 60, KH, NH);
    pack_b_f16<<<(NH * NB + TPB - 1) / TPB, TPB, 0, stream>>>(W1, W1P, NH, NH, NB);
    pack_b_f16<<<(NB * NP + TPB - 1) / TPB, TPB, 0, stream>>>(W2, W2P, NB, NB, NP);

    // GEMM1: h = relu(X @ Wh + b_h)   [4096x64]x[64x512]
    {
        int waves = (ROWS / 16) * (NH / 16);           // 8192
        gemm_bias_relu_f16<<<waves * 32 / TPB, TPB, 0, stream>>>(
            Xp, WhP, b_h, Hb, ROWS, KH, NH);
    }
    // GEMM2: t = selu(h @ W1 + b1)    [4096x512]x[512x160]
    {
        int waves = (ROWS / 16) * (NB / 16);           // 2560
        gemm_bias_selu_f16<<<waves * 32 / TPB, TPB, 0, stream>>>(
            Hb, W1P, b1, Tb, ROWS, NH, NB);
    }
    // GEMM3 fused with sigmoid + BCE + Wd-reduction.
    {
        int waves = (ROWS / 16) * GROUPS;              // 49152
        gemm_sigmoid_bce<<<waves * 32 / TPB, TPB, 0, stream>>>(
            Tb, W2P, b2, images, out, ROWS, NB, NP);
    }
}
